// SelfAttentionBlock_47090021434067
// MI455X (gfx1250) — matrix operand order repaired and checked by reading, not compiled, not run on hardware
//
#include <hip/hip_runtime.h>
#include <hip/hip_bf16.h>
#include <math.h>

// Problem constants (fixed by the reference)
#define B_ 8
#define T_ 2048
#define E_ 1024
#define H_ 128
#define SCALE_ 0.088388347648318447f  // 1/sqrt(128)
#define NEG_BIG -3.0e38f

typedef __attribute__((ext_vector_type(16))) _Float16 v16h;
typedef __attribute__((ext_vector_type(8)))  _Float16 v8h;
typedef __attribute__((ext_vector_type(8)))  float    v8f;
typedef __attribute__((ext_vector_type(4)))  float    v4f;

__device__ __forceinline__ v8f wmma16(v16h a, v16h b, v8f c) {
  // D(f32 16x16) = A(f16 16x32) * B(f16 32x16) + C
  return __builtin_amdgcn_wmma_f32_16x16x32_f16(
      /*neg_a=*/false, a, /*neg_b=*/false, b,
      /*c_mod=*/(short)0, c, /*reuse_a=*/false, /*reuse_b=*/false);
}

// Machine-scheduler fence: nothing may be reordered across this point.
__device__ __forceinline__ void sched_fence() {
  __builtin_amdgcn_sched_barrier(0);
}

__device__ __forceinline__ v16h cat8(v8h a, v8h b) {
  v16h r;
#pragma unroll
  for (int i = 0; i < 8; ++i) { r[i] = a[i]; r[8 + i] = b[i]; }
  return r;
}

__device__ __forceinline__ v8f zero8() {
  v8f z;
#pragma unroll
  for (int i = 0; i < 8; ++i) z[i] = 0.0f;
  return z;
}

__device__ __forceinline__ v8h ldg8h(const _Float16* p) {
  return *reinterpret_cast<const v8h*>(p);  // global_load_b128
}

// ---------------------------------------------------------------------------
// Kernel 1: q/k/v = x @ W{q,k,v}, outputs f16; V stored transposed [B,H,T].
// Grid: (M/128, 3). Block: 256 threads = 8 waves; wave w owns rows [w*16,w*16+16).
// Double-buffered LDS staging of W k-slices; B-fragments preloaded (enforced
// with a sched barrier) so the 8 WMMAs per k-step issue back-to-back.
// ---------------------------------------------------------------------------
__global__ __launch_bounds__(256) void qkv_proj_kernel(
    const float* __restrict__ x,
    const float* __restrict__ Wq, const float* __restrict__ Wk,
    const float* __restrict__ Wv,
    _Float16* __restrict__ q16, _Float16* __restrict__ k16,
    _Float16* __restrict__ vt16) {
  __shared__ _Float16 ldsWt[2][H_][40];  // W k-slice transposed; 80B rows keep 16B align

  const int mode = blockIdx.y;  // 0 = Q, 1 = K, 2 = V
  const float* __restrict__ W = (mode == 0) ? Wq : ((mode == 1) ? Wk : Wv);

  const int tid  = threadIdx.x;
  const int lane = tid & 31;
  const int wv   = tid >> 5;        // wave 0..7
  const int hi   = lane >> 4;       // half-wave selector
  const int ln   = lane & 15;
  const int rowBase = blockIdx.x * 128 + wv * 16;

  v8f acc[8];
#pragma unroll
  for (int t = 0; t < 8; ++t) acc[t] = zero8();

  // Pipeline: W slice for k-step 0 preloaded into registers.
  const int wr  = tid >> 3;         // 0..31 (k row within slice)
  const int wc0 = (tid & 7) * 16;   // 16 contiguous cols per thread
  v4f w0, w1, w2, w3;
  {
    const v4f* wp = reinterpret_cast<const v4f*>(W + (size_t)wr * H_ + wc0);
    w0 = wp[0]; w1 = wp[1]; w2 = wp[2]; w3 = wp[3];
  }

  int buf = 0;
  for (int k0 = 0; k0 < E_; k0 += 32, buf ^= 1) {
    // Commit the staged slice (f32->f16, transposed) into this iteration's buffer.
#pragma unroll
    for (int i = 0; i < 4; ++i) {
      ldsWt[buf][wc0 + 0  + i][wr] = (_Float16)w0[i];
      ldsWt[buf][wc0 + 4  + i][wr] = (_Float16)w1[i];
      ldsWt[buf][wc0 + 8  + i][wr] = (_Float16)w2[i];
      ldsWt[buf][wc0 + 12 + i][wr] = (_Float16)w3[i];
    }
    __syncthreads();

    // Prefetch next W slice into registers (overlaps the WMMA stream below).
    if (k0 + 32 < E_) {
      const v4f* wp =
          reinterpret_cast<const v4f*>(W + (size_t)(k0 + 32 + wr) * H_ + wc0);
      w0 = wp[0]; w1 = wp[1]; w2 = wp[2]; w3 = wp[3];
    }

    // A fragment: 16x32 f16 slice of x (converted f32->f16 in registers).
    // Lane layout: row = ln; lanes 0-15 hold K {0..7,16..23}, lanes 16-31 {8..15,24..31}.
    const size_t xoff = (size_t)(rowBase + ln) * E_ + k0 + hi * 8;
    const v4f* xa = reinterpret_cast<const v4f*>(x + xoff);
    const v4f* xb = reinterpret_cast<const v4f*>(x + xoff + 16);
    v4f a0 = xa[0], a1 = xa[1], a2 = xb[0], a3 = xb[1];
    v16h af;
#pragma unroll
    for (int i = 0; i < 4; ++i) {
      af[i]      = (_Float16)a0[i];
      af[4 + i]  = (_Float16)a1[i];
      af[8 + i]  = (_Float16)a2[i];
      af[12 + i] = (_Float16)a3[i];
    }

    // Preload all 8 B fragments; fence so the scheduler cannot sink the
    // ds_loads back down next to their WMMA consumers.
    v16h bf[8];
#pragma unroll
    for (int nt = 0; nt < 8; ++nt) {
      const _Float16* bp = &ldsWt[buf][nt * 16 + ln][hi * 8];
      bf[nt] = cat8(*reinterpret_cast<const v8h*>(bp),
                    *reinterpret_cast<const v8h*>(bp + 16));
    }
    sched_fence();
#pragma unroll
    for (int nt = 0; nt < 8; ++nt) acc[nt] = wmma16(af, bf[nt], acc[nt]);
    sched_fence();
    // No trailing barrier: next iteration writes the other LDS buffer, and the
    // single barrier above guarantees the two-iteration reuse distance.
  }

  // Store. C-layout: VGPR r, lanes 0-15 -> row r, lanes 16-31 -> row r+8; col = ln.
  if (mode == 2) {
    // V transposed: vt16[(b*H + h)*T + t]
#pragma unroll
    for (int nt = 0; nt < 8; ++nt) {
      const int h = nt * 16 + ln;
#pragma unroll
      for (int r = 0; r < 8; ++r) {
        const int gr = rowBase + r + hi * 8;
        const int bb = gr / T_;
        const int tl = gr % T_;
        vt16[((size_t)bb * H_ + h) * T_ + tl] = (_Float16)acc[nt][r];
      }
    }
  } else {
    _Float16* __restrict__ dst = (mode == 0) ? q16 : k16;
#pragma unroll
    for (int nt = 0; nt < 8; ++nt) {
      const int h = nt * 16 + ln;
#pragma unroll
      for (int r = 0; r < 8; ++r) {
        dst[(size_t)(rowBase + r + hi * 8) * H_ + h] = (_Float16)acc[nt][r];
      }
    }
  }
}

// ---------------------------------------------------------------------------
// Kernel 2: flash attention. Each wave owns a 16-row Q tile; loops over 32-key
// blocks up to the causal frontier with online softmax. Software-pipelined:
// K-frags load -> S WMMA burst -> V-frag loads issue -> softmax VALU overlaps
// the V loads -> PV WMMA burst.
// Grid: (B*T/128). Block: 256 threads = 8 independent waves.
// ---------------------------------------------------------------------------
__global__ __launch_bounds__(256) void attn_kernel(
    const _Float16* __restrict__ q16, const _Float16* __restrict__ k16,
    const _Float16* __restrict__ vt16, float* __restrict__ out) {
  __shared__ _Float16 ldsP[8][16][40];  // per-wave P tile (16 rows x 32 keys)

  const int tid  = threadIdx.x;
  const int lane = tid & 31;
  const int wv   = tid >> 5;
  const int hi   = lane >> 4;
  const int ln   = lane & 15;

  const int m0 = blockIdx.x * 128 + wv * 16;  // global query-row base
  const int bb = m0 / T_;                     // batch (block never straddles)
  const int t0 = m0 % T_;                     // in-batch query position base

  const _Float16* __restrict__ kbase = k16  + (size_t)bb * T_ * H_;
  const _Float16* __restrict__ vbase = vt16 + (size_t)bb * H_ * T_;

  // Q as four 16x32 A-fragments (K = H = 128 split in 4).
  v16h qf[4];
  {
    const _Float16* qr = q16 + (size_t)(m0 + ln) * H_;
#pragma unroll
    for (int kk = 0; kk < 4; ++kk) {
      const int kb = kk * 32 + hi * 8;
      qf[kk] = cat8(ldg8h(qr + kb), ldg8h(qr + kb + 16));
    }
  }

  v8f acc[8];
#pragma unroll
  for (int t = 0; t < 8; ++t) acc[t] = zero8();
  float rmax[8], rsum[8];
#pragma unroll
  for (int r = 0; r < 8; ++r) { rmax[r] = NEG_BIG; rsum[r] = 0.0f; }

  // Causal key blocks: every processed block satisfies j <= m for all its rows.
  for (int j = 0; j < t0 + 16; j += 32) {
    // --- Load all 8 K-fragments (both 16-key tiles) up front ---
    v16h kf[2][4];
#pragma unroll
    for (int nt = 0; nt < 2; ++nt) {
#pragma unroll
      for (int kk = 0; kk < 4; ++kk) {
        // B(kdim=h, n=key): lane col = key, contiguous along h in K[T,H].
        const _Float16* kr =
            kbase + (size_t)(j + nt * 16 + ln) * H_ + kk * 32 + hi * 8;
        kf[nt][kk] = cat8(ldg8h(kr), ldg8h(kr + 16));
      }
    }
    sched_fence();

    // --- S = Q K^T : 8 WMMAs in one burst ---
    v8f sacc[2];
#pragma unroll
    for (int nt = 0; nt < 2; ++nt) {
      v8f c = zero8();
#pragma unroll
      for (int kk = 0; kk < 4; ++kk) c = wmma16(qf[kk], kf[nt][kk], c);
      sacc[nt] = c;
    }
    sched_fence();

    // --- Issue V-fragment loads now; they complete under the softmax below ---
    v16h vf[8];
#pragma unroll
    for (int nt = 0; nt < 8; ++nt) {
      // B(kdim=key, n=h): V^T[b][h][key] is contiguous along keys.
      const _Float16* vr =
          vbase + (size_t)(nt * 16 + ln) * T_ + j + hi * 8;
      vf[nt] = cat8(ldg8h(vr), ldg8h(vr + 16));
    }
    sched_fence();

    // --- scale + causal mask + online softmax (rows span 16-lane groups) ---
#pragma unroll
    for (int r = 0; r < 8; ++r) {
      const int m  = t0 + r + hi * 8;
      const int n0 = j + ln;
      const int n1 = j + 16 + ln;
      float s0 = sacc[0][r] * SCALE_; if (n0 > m) s0 = NEG_BIG;
      float s1 = sacc[1][r] * SCALE_; if (n1 > m) s1 = NEG_BIG;

      float mx = fmaxf(s0, s1);
      mx = fmaxf(mx, __shfl_xor(mx, 1, 32));
      mx = fmaxf(mx, __shfl_xor(mx, 2, 32));
      mx = fmaxf(mx, __shfl_xor(mx, 4, 32));
      mx = fmaxf(mx, __shfl_xor(mx, 8, 32));

      const float nm    = fmaxf(rmax[r], mx);
      const float alpha = __expf(rmax[r] - nm);
      const float e0    = __expf(s0 - nm);
      const float e1    = __expf(s1 - nm);

      float ps = e0 + e1;
      ps += __shfl_xor(ps, 1, 32);
      ps += __shfl_xor(ps, 2, 32);
      ps += __shfl_xor(ps, 4, 32);
      ps += __shfl_xor(ps, 8, 32);

      rsum[r] = rsum[r] * alpha + ps;
      rmax[r] = nm;
#pragma unroll
      for (int t = 0; t < 8; ++t) acc[t][r] *= alpha;

      // Stash P (f16) for layout transpose C->A via LDS.
      ldsP[wv][r + hi * 8][ln]      = (_Float16)e0;
      ldsP[wv][r + hi * 8][16 + ln] = (_Float16)e1;
    }

    // --- O += P V : V-frags already resident; 8 WMMAs back-to-back ---
    v16h pf;
    {
      const _Float16* pr = &ldsP[wv][ln][hi * 8];
      pf = cat8(*reinterpret_cast<const v8h*>(pr),
                *reinterpret_cast<const v8h*>(pr + 16));
    }
    sched_fence();
#pragma unroll
    for (int nt = 0; nt < 8; ++nt) acc[nt] = wmma16(pf, vf[nt], acc[nt]);
    sched_fence();
  }

  // --- normalize and write out (f32, [B,T,H]) ---
  float inv[8];
#pragma unroll
  for (int r = 0; r < 8; ++r) inv[r] = 1.0f / rsum[r];
#pragma unroll
  for (int nt = 0; nt < 8; ++nt) {
    const int h = nt * 16 + ln;
#pragma unroll
    for (int r = 0; r < 8; ++r) {
      const int gr = m0 + r + hi * 8;
      out[(size_t)gr * H_ + h] = acc[nt][r] * inv[r];
    }
  }
}

// ---------------------------------------------------------------------------
extern "C" void kernel_launch(void* const* d_in, const int* in_sizes, int n_in,
                              void* d_out, int out_size, void* d_ws,
                              size_t ws_size, hipStream_t stream) {
  (void)in_sizes; (void)n_in; (void)out_size; (void)ws_size;
  const float* x  = (const float*)d_in[0];
  // d_in[1] (mask) unused: causal mask applied analytically.
  const float* Wk = (const float*)d_in[2];
  const float* Wq = (const float*)d_in[3];
  const float* Wv = (const float*)d_in[4];

  // Workspace: q (4MB) | k (4MB) | v^T (4MB), all f16.
  _Float16* q16  = (_Float16*)d_ws;
  _Float16* k16  = q16 + (size_t)B_ * T_ * H_;
  _Float16* vt16 = k16 + (size_t)B_ * T_ * H_;

  dim3 g1((B_ * T_) / 128, 3), blk(256);
  qkv_proj_kernel<<<g1, blk, 0, stream>>>(x, Wq, Wk, Wv, q16, k16, vt16);

  dim3 g2((B_ * T_) / 128);
  attn_kernel<<<g2, blk, 0, stream>>>(q16, k16, vt16, (float*)d_out);
}
